// TL_43671227465816
// MI455X (gfx1250) — matrix operand order repaired and checked by reading, not compiled, not run on hardware
//
#include <hip/hip_runtime.h>
#include <hip/hip_bf16.h>

typedef __bf16 v16bf __attribute__((ext_vector_type(16)));
typedef float  v8f   __attribute__((ext_vector_type(8)));

#define N0C 524288
#define N1C 131072
#define BC  512
#define HC  128
#define EPSC 1e-5f

// ---------------------------------------------------------------------------
// zero helper (float4 per thread)
// ---------------------------------------------------------------------------
__global__ __launch_bounds__(256) void zero4_kernel(float4* __restrict__ p, int n4) {
    int i = blockIdx.x * 256 + threadIdx.x;
    if (i < n4) p[i] = make_float4(0.f, 0.f, 0.f, 0.f);
}

// ---------------------------------------------------------------------------
// weight prep: f32 [K=128][N=128] row-major  ->  bf16 [N=128][K=128] (K contig)
// ---------------------------------------------------------------------------
__global__ __launch_bounds__(256) void wprep_kernel(const float* __restrict__ W,
                                                    __bf16* __restrict__ Wt) {
    int idx = blockIdx.x * 256 + threadIdx.x;   // 16384 elements
    int k = idx >> 7;
    int n = idx & 127;
    Wt[n * 128 + k] = (__bf16)W[idx];
}

// ---------------------------------------------------------------------------
// fused scatter of node_features into p1 (by parent1) and g0 (by tree0)
// one thread = one node x 4 columns (float4)
// ---------------------------------------------------------------------------
__global__ __launch_bounds__(256) void scatter1_kernel(const float* __restrict__ x,
                                                       const int* __restrict__ parent1,
                                                       const int* __restrict__ tree0,
                                                       float* __restrict__ p1,
                                                       float* __restrict__ g0) {
    int idx  = blockIdx.x * 256 + threadIdx.x;
    int node = idx >> 5;            // 32 groups of 4 cols per node
    int grp  = idx & 31;
    if (node >= N0C) return;
    float4 v = ((const float4*)x)[node * 32 + grp];
    int p = parent1[node];
    int t = tree0[node];
    float* pd = p1 + p * 128 + grp * 4;
    float* gd = g0 + t * 128 + grp * 4;
    atomicAdd(pd + 0, v.x); atomicAdd(pd + 1, v.y);
    atomicAdd(pd + 2, v.z); atomicAdd(pd + 3, v.w);
    atomicAdd(gd + 0, v.x); atomicAdd(gd + 1, v.y);
    atomicAdd(gd + 2, v.z); atomicAdd(gd + 3, v.w);
}

// ---------------------------------------------------------------------------
// fused affine+ReLU scatter of h1 into p2 (parent2) and g1 (tree1)
// ---------------------------------------------------------------------------
__global__ __launch_bounds__(256) void scatter2_kernel(const float* __restrict__ t2,
                                                       const int* __restrict__ parent2,
                                                       const int* __restrict__ tree1,
                                                       const float* __restrict__ scale,
                                                       const float* __restrict__ shift,
                                                       float* __restrict__ p2,
                                                       float* __restrict__ g1) {
    int idx  = blockIdx.x * 256 + threadIdx.x;
    int node = idx >> 5;
    int grp  = idx & 31;
    if (node >= N1C) return;
    float4 v = ((const float4*)t2)[node * 32 + grp];
    int k = grp * 4;
    v.x = fmaxf(v.x * scale[k + 0] + shift[k + 0], 0.f);
    v.y = fmaxf(v.y * scale[k + 1] + shift[k + 1], 0.f);
    v.z = fmaxf(v.z * scale[k + 2] + shift[k + 2], 0.f);
    v.w = fmaxf(v.w * scale[k + 3] + shift[k + 3], 0.f);
    int p = parent2[node];
    int t = tree1[node];
    float* pd = p2 + p * 128 + k;
    float* gd = g1 + t * 128 + k;
    atomicAdd(pd + 0, v.x); atomicAdd(pd + 1, v.y);
    atomicAdd(pd + 2, v.z); atomicAdd(pd + 3, v.w);
    atomicAdd(gd + 0, v.x); atomicAdd(gd + 1, v.y);
    atomicAdd(gd + 2, v.z); atomicAdd(gd + 3, v.w);
}

// ---------------------------------------------------------------------------
// WMMA GEMM:  Out[N,128] = act(A[N,128]) @ Wt^T + bias
// Wt is bf16 [128 out][128 in] with K contiguous per output column.
// AFFINE: A element pre-op  x -> relu(x*scale[k]+shift[k])  (BN folded in)
// Block = 256 threads (8 waves); each wave -> 16 rows x 128 cols;
// block -> 128 rows. 32 v_wmma_f32_16x16x32_bf16 per wave.
//
// Lane layouts used (A/B in contiguous k order, not the ISA table order; D as in CDNA5 ISA):
//   A 16x32 bf16: lane l (l<16): M=l, elems 0..15 = K 0..15 contig
//                 lane l (>=16): M=l-16, elems 0..15 = K 16..31 contig
//   B 32x16 bf16: lanes 0-15: N=lane, K 0..15 contig; lanes 16-31: N=lane-16, K 16..31
//   D 16x16 f32 : vgpr v: lanes 0-15 -> M=v, N=lane; lanes 16-31 -> M=8+v, N=lane-16
// ---------------------------------------------------------------------------
template <bool AFFINE>
__global__ __launch_bounds__(256) void gemm128_kernel(const float* __restrict__ A,
                                                      const __bf16* __restrict__ Bt,
                                                      const float* __restrict__ bias,
                                                      const float* __restrict__ scale,
                                                      const float* __restrict__ shift,
                                                      float* __restrict__ Out) {
    __shared__ __align__(32) __bf16 sB[128 * 128];

    const int tid = threadIdx.x;
    // stage bf16 weights (32 KB) into LDS
    {
        const uint4* src = (const uint4*)Bt;
        uint4* dst = (uint4*)sB;
#pragma unroll
        for (int i = 0; i < 8; ++i) dst[tid + 256 * i] = src[tid + 256 * i];
    }
    __syncthreads();

    const int wave  = tid >> 5;
    const int lane  = tid & 31;
    const int lrow  = lane & 15;
    const int khalf = lane >> 4;
    const int rowBlock = blockIdx.x * 128 + wave * 16;
    const float* arow = A + (rowBlock + lrow) * 128;

    // build the 4 A fragments (K = s*32 .. s*32+31)
    v16bf afrag[4];
#pragma unroll
    for (int s = 0; s < 4; ++s) {
#pragma unroll
        for (int h = 0; h < 2; ++h) {            // elems h*8 .. h*8+7
            const int kb = s * 32 + khalf * 16 + h * 8;
#pragma unroll
            for (int q = 0; q < 2; ++q) {        // two float4 loads per 8 elems
                float4 f = *(const float4*)(arow + kb + q * 4);
                float x0 = f.x, x1 = f.y, x2 = f.z, x3 = f.w;
                if (AFFINE) {
                    const int k = kb + q * 4;
                    x0 = fmaxf(x0 * scale[k + 0] + shift[k + 0], 0.f);
                    x1 = fmaxf(x1 * scale[k + 1] + shift[k + 1], 0.f);
                    x2 = fmaxf(x2 * scale[k + 2] + shift[k + 2], 0.f);
                    x3 = fmaxf(x3 * scale[k + 3] + shift[k + 3], 0.f);
                }
                const int e = h * 8 + q * 4;
                afrag[s][e + 0] = (__bf16)x0;
                afrag[s][e + 1] = (__bf16)x1;
                afrag[s][e + 2] = (__bf16)x2;
                afrag[s][e + 3] = (__bf16)x3;
            }
        }
    }

    float* orow = Out + rowBlock * 128;
#pragma unroll
    for (int n = 0; n < 8; ++n) {
        const int col = n * 16 + lrow;
        v8f c = {};
#pragma unroll
        for (int s = 0; s < 4; ++s) {
            const v16bf bfrag =
                *(const v16bf*)(sB + col * 128 + s * 32 + khalf * 16);
            c = __builtin_amdgcn_wmma_f32_16x16x32_bf16(
                    false, afrag[s], false, bfrag, (short)0, c, false, false);
        }
        const float bs = bias[col];
#pragma unroll
        for (int v = 0; v < 8; ++v)
            orow[(khalf * 8 + v) * 128 + col] = c[v] + bs;
    }
}

// ---------------------------------------------------------------------------
// per-column sum / sum-of-squares (BN batch stats), LDS partial + atomics
// block = 256 threads, handles rowsPerBlock rows of 128 cols
// ---------------------------------------------------------------------------
__global__ __launch_bounds__(256) void colstats_kernel(const float* __restrict__ X,
                                                       int rowsPerBlock,
                                                       float* __restrict__ sums,
                                                       float* __restrict__ sqs) {
    __shared__ float sh[512];
    const int tid  = threadIdx.x;
    const int col  = tid & 127;
    const int half = tid >> 7;
    const int row0 = blockIdx.x * rowsPerBlock;
    float s = 0.f, q = 0.f;
    for (int r = half; r < rowsPerBlock; r += 2) {
        float x = X[(row0 + r) * 128 + col];
        s += x;
        q += x * x;
    }
    sh[tid] = s;
    sh[256 + tid] = q;
    __syncthreads();
    if (tid < 128) {
        s = sh[tid] + sh[tid + 128];
        q = sh[256 + tid] + sh[256 + tid + 128];
        atomicAdd(&sums[col], s);
        atomicAdd(&sqs[col], q);
    }
}

// ---------------------------------------------------------------------------
// fold BN stats + gamma/beta into per-column scale/shift
// ---------------------------------------------------------------------------
__global__ __launch_bounds__(128) void finalize_kernel(const float* __restrict__ sums,
                                                       const float* __restrict__ sqs,
                                                       const float* __restrict__ g,
                                                       const float* __restrict__ b,
                                                       float invN,
                                                       float* __restrict__ scale,
                                                       float* __restrict__ shift) {
    int c = threadIdx.x;
    float m   = sums[c] * invN;
    float var = sqs[c] * invN - m * m;
    float rs  = rsqrtf(var + EPSC);
    float sc  = g[c] * rs;
    scale[c] = sc;
    shift[c] = b[c] - m * sc;
}

// ---------------------------------------------------------------------------
// per-tree: net = [g0 | g1 | relu(affine(t4))] ; softmax(net @ predW + predb)
// one block (128 threads) per tree
// ---------------------------------------------------------------------------
__global__ __launch_bounds__(128) void final_kernel(const float* __restrict__ g0,
                                                    const float* __restrict__ g1,
                                                    const float* __restrict__ t4,
                                                    const float* __restrict__ sc4,
                                                    const float* __restrict__ sh4,
                                                    const float* __restrict__ predW,
                                                    const float* __restrict__ predb,
                                                    float* __restrict__ out) {
    __shared__ float net[384];
    __shared__ float acc[128][10];
    const int b = blockIdx.x;
    const int t = threadIdx.x;
    net[t]       = g0[b * 128 + t];
    net[128 + t] = g1[b * 128 + t];
    net[256 + t] = fmaxf(t4[b * 128 + t] * sc4[t] + sh4[t], 0.f);
    __syncthreads();

    float a[10];
#pragma unroll
    for (int c = 0; c < 10; ++c) a[c] = 0.f;
    for (int j = t; j < 384; j += 128) {
        float x = net[j];
#pragma unroll
        for (int c = 0; c < 10; ++c) a[c] += x * predW[j * 10 + c];
    }
#pragma unroll
    for (int c = 0; c < 10; ++c) acc[t][c] = a[c];
    __syncthreads();
    for (int stride = 64; stride > 0; stride >>= 1) {
        if (t < stride) {
#pragma unroll
            for (int c = 0; c < 10; ++c) acc[t][c] += acc[t + stride][c];
        }
        __syncthreads();
    }
    if (t == 0) {
        float l[10], m = -1e30f, s = 0.f;
#pragma unroll
        for (int c = 0; c < 10; ++c) {
            l[c] = acc[0][c] + predb[c];
            m = fmaxf(m, l[c]);
        }
#pragma unroll
        for (int c = 0; c < 10; ++c) {
            l[c] = __expf(l[c] - m);
            s += l[c];
        }
        float inv = 1.f / s;
#pragma unroll
        for (int c = 0; c < 10; ++c) out[b * 10 + c] = l[c] * inv;
    }
}

// ---------------------------------------------------------------------------
extern "C" void kernel_launch(void* const* d_in, const int* in_sizes, int n_in,
                              void* d_out, int out_size, void* d_ws, size_t ws_size,
                              hipStream_t stream) {
    const float* nodef   = (const float*)d_in[0];
    const int*   parent1 = (const int*)d_in[1];
    const int*   parent2 = (const int*)d_in[2];
    const int*   tree0   = (const int*)d_in[3];
    const int*   tree1   = (const int*)d_in[4];
    const float* m1W1 = (const float*)d_in[5];
    const float* m1b1 = (const float*)d_in[6];
    const float* m1bng = (const float*)d_in[7];
    const float* m1bnb = (const float*)d_in[8];
    const float* m1W2 = (const float*)d_in[9];
    const float* m1b2 = (const float*)d_in[10];
    const float* bn1g = (const float*)d_in[11];
    const float* bn1b = (const float*)d_in[12];
    const float* m2W1 = (const float*)d_in[13];
    const float* m2b1 = (const float*)d_in[14];
    const float* m2bng = (const float*)d_in[15];
    const float* m2bnb = (const float*)d_in[16];
    const float* m2W2 = (const float*)d_in[17];
    const float* m2b2 = (const float*)d_in[18];
    const float* bn2g = (const float*)d_in[19];
    const float* bn2b = (const float*)d_in[20];
    const float* predW = (const float*)d_in[21];
    const float* predb = (const float*)d_in[22];
    float* out = (float*)d_out;

    // ---- workspace layout ----
    float* bufA  = (float*)d_ws;           // [N1*128] : p1, then reused for t2
    float* bufB  = bufA + (size_t)N1C * 128;  // [N1*128] : t1
    float* g0    = bufB + (size_t)N1C * 128;  // [512*128]
    float* g1    = g0 + BC * 128;
    float* p2    = g1 + BC * 128;
    float* t3    = p2 + BC * 128;
    float* t4    = t3 + BC * 128;
    float* stats = t4 + BC * 128;          // 8*128 : (sum,sq) x 4 stages
    float* scsh  = stats + 1024;           // 8*128 : (scale,shift) x 4 stages
    __bf16* Wt1 = (__bf16*)(scsh + 1024);
    __bf16* Wt2 = Wt1 + 16384;
    __bf16* Wt3 = Wt2 + 16384;
    __bf16* Wt4 = Wt3 + 16384;

    // ---- zero accumulators ----
    zero4_kernel<<<(N1C * 128 / 4 + 255) / 256, 256, 0, stream>>>((float4*)bufA, N1C * 128 / 4);
    zero4_kernel<<<(3 * BC * 128 / 4 + 255) / 256, 256, 0, stream>>>((float4*)g0, 3 * BC * 128 / 4);
    zero4_kernel<<<1, 256, 0, stream>>>((float4*)stats, 1024 / 4);

    // ---- weight prep (f32 -> bf16, K-contiguous) ----
    wprep_kernel<<<64, 256, 0, stream>>>(m1W1, Wt1);
    wprep_kernel<<<64, 256, 0, stream>>>(m1W2, Wt2);
    wprep_kernel<<<64, 256, 0, stream>>>(m2W1, Wt3);
    wprep_kernel<<<64, 256, 0, stream>>>(m2W2, Wt4);

    // ---- layer-1 aggregation: h0 -> p1 (parent1), g0 (tree0) ----
    scatter1_kernel<<<(size_t)N0C * 32 / 256, 256, 0, stream>>>(nodef, parent1, tree0, bufA, g0);

    // ---- MLP1: t1 = p1 @ W1 + b1 ; BN stats ; fold ----
    gemm128_kernel<false><<<N1C / 128, 256, 0, stream>>>(bufA, Wt1, m1b1, nullptr, nullptr, bufB);
    colstats_kernel<<<N1C / 128, 256, 0, stream>>>(bufB, 128, stats + 0, stats + 128);
    finalize_kernel<<<1, 128, 0, stream>>>(stats + 0, stats + 128, m1bng, m1bnb,
                                           1.f / N1C, scsh + 0, scsh + 128);
    // t2 = relu(bn(t1)) @ W2 + b2 ; BN stats ; fold (outer bn1)
    gemm128_kernel<true><<<N1C / 128, 256, 0, stream>>>(bufB, Wt2, m1b2, scsh + 0, scsh + 128, bufA);
    colstats_kernel<<<N1C / 128, 256, 0, stream>>>(bufA, 128, stats + 256, stats + 384);
    finalize_kernel<<<1, 128, 0, stream>>>(stats + 256, stats + 384, bn1g, bn1b,
                                           1.f / N1C, scsh + 256, scsh + 384);

    // ---- layer-2 aggregation: h1 = relu(bn1(t2)) -> p2 (parent2), g1 (tree1) ----
    scatter2_kernel<<<(size_t)N1C * 32 / 256, 256, 0, stream>>>(bufA, parent2, tree1,
                                                                scsh + 256, scsh + 384, p2, g1);

    // ---- MLP2 on [512,128] ----
    gemm128_kernel<false><<<BC / 128, 256, 0, stream>>>(p2, Wt3, m2b1, nullptr, nullptr, t3);
    colstats_kernel<<<BC / 128, 256, 0, stream>>>(t3, 128, stats + 512, stats + 640);
    finalize_kernel<<<1, 128, 0, stream>>>(stats + 512, stats + 640, m2bng, m2bnb,
                                           1.f / BC, scsh + 512, scsh + 640);
    gemm128_kernel<true><<<BC / 128, 256, 0, stream>>>(t3, Wt4, m2b2, scsh + 512, scsh + 640, t4);
    colstats_kernel<<<BC / 128, 256, 0, stream>>>(t4, 128, stats + 768, stats + 896);
    finalize_kernel<<<1, 128, 0, stream>>>(stats + 768, stats + 896, bn2g, bn2b,
                                           1.f / BC, scsh + 768, scsh + 896);

    // ---- pooled concat + predictor + softmax ----
    final_kernel<<<BC, 128, 0, stream>>>(g0, g1, t4, scsh + 768, scsh + 896,
                                         predW, predb, out);
}
